// GnS_58591943852540
// MI455X (gfx1250) — matrix operand order repaired and checked by reading, not compiled, not run on hardware
//
#include <hip/hip_runtime.h>
#include <hip/hip_bf16.h>

// ---------------------------------------------------------------------------
// GraphDTA GINConvNet forward for MI455X (gfx1250, wave32, WMMA).
//
// Param leaf order assumption (insertion order of setup_inputs()):
//   0 xd[40960,55] 1 edge_index[2,163840] 2 batch[40960] 3 xt[1024,1000] 4 y[1024]
//   5.. per GIN layer li (base=5+8*li): l1.W,l1.b,l2.W,l2.b,bn.g,bn.b,bn.m,bn.v
//   45 fc1_xd.W[32,128] 46 fc1_xd.b 47 emb_xt[26,128] 48 conv.W[32,1000,8]
//   49 conv.b[32] 50 fc1_xt.W[3872,128] 51 fc1_xt.b 52..57 cls{0,1,2}.{W,b}
// ---------------------------------------------------------------------------

typedef __attribute__((ext_vector_type(16))) _Float16 v16h;
typedef __attribute__((ext_vector_type(8)))  _Float16 v8h;
typedef __attribute__((ext_vector_type(8)))  float    v8f;

#define WMMA16(a, b, c) \
  __builtin_amdgcn_wmma_f32_16x16x32_f16(false, (a), false, (b), (short)0, (c), false, false)

static __device__ __forceinline__ v8f zero8() {
  v8f z = {0.f, 0.f, 0.f, 0.f, 0.f, 0.f, 0.f, 0.f};
  return z;
}

// A fragment (16x32 f16) from row-major f32, K-range fully in bounds -> 4x b128.
static __device__ __forceinline__ v16h load_A_f32x32(const float* __restrict__ s, int ld,
                                                     int m0, int k0) {
  int lane = threadIdx.x & 31;
  int m = m0 + (lane & 15);
  int kb = (lane >> 4) * 8;
  const float* p = s + (size_t)m * ld + k0 + kb;
  float4 x0 = *(const float4*)(p);
  float4 x1 = *(const float4*)(p + 4);
  float4 x2 = *(const float4*)(p + 16);
  float4 x3 = *(const float4*)(p + 20);
  v16h a;
  a[0] = (_Float16)x0.x; a[1] = (_Float16)x0.y; a[2] = (_Float16)x0.z; a[3] = (_Float16)x0.w;
  a[4] = (_Float16)x1.x; a[5] = (_Float16)x1.y; a[6] = (_Float16)x1.z; a[7] = (_Float16)x1.w;
  a[8] = (_Float16)x2.x; a[9] = (_Float16)x2.y; a[10] = (_Float16)x2.z; a[11] = (_Float16)x2.w;
  a[12] = (_Float16)x3.x; a[13] = (_Float16)x3.y; a[14] = (_Float16)x3.z; a[15] = (_Float16)x3.w;
  return a;
}

// A fragment from row-major f16 (LDS or global), vectorized (16B aligned runs).
static __device__ __forceinline__ v16h load_A_f16v(const _Float16* __restrict__ s, int ld,
                                                   int m0, int k0) {
  int lane = threadIdx.x & 31;
  int m = m0 + (lane & 15);
  int kb = (lane >> 4) * 8;
  const _Float16* p = s + (size_t)m * ld + k0 + kb;
  v8h lo = *(const v8h*)(p);
  v8h hi = *(const v8h*)(p + 16);
  v16h a;
#pragma unroll
  for (int e = 0; e < 8; ++e) { a[e] = lo[e]; a[e + 8] = hi[e]; }
  return a;
}

// B fragment from fragment-linear packed f16 weights: layout [ks][n][kb][e],
// 16 contiguous halves per (lane,kb) -> two b128 loads.
static __device__ __forceinline__ v16h load_B_pack(const _Float16* __restrict__ Bp, int N,
                                                   int ks, int n0) {
  int lane = threadIdx.x & 31;
  int n = n0 + (lane & 15);
  int kb = lane >> 4;
  const _Float16* p = Bp + (((size_t)ks * N + n) * 2 + kb) * 16;
  v8h lo = *(const v8h*)(p);
  v8h hi = *(const v8h*)(p + 8);
  v16h b;
#pragma unroll
  for (int e = 0; e < 8; ++e) { b[e] = lo[e]; b[e + 8] = hi[e]; }
  return b;
}

// B fragment (32x16 f16) from row-major f16 in LDS (strided ds reads).
static __device__ __forceinline__ v16h load_B_f16(const _Float16* __restrict__ s, int ld,
                                                  int k0, int n0) {
  int lane = threadIdx.x & 31;
  int n = n0 + (lane & 15);
  int kb = (lane >> 4) * 8;
  v16h b;
#pragma unroll
  for (int e = 0; e < 16; ++e) {
    b[e] = s[(size_t)(k0 + kb + e + (e >= 8 ? 8 : 0)) * ld + n];
  }
  return b;
}

// -------------------------- utility kernels --------------------------------

__global__ void gdta_zero(float* __restrict__ p, int n) {
  int i = blockIdx.x * blockDim.x + threadIdx.x;
  if (i < n) p[i] = 0.f;
}

// Repack row-major f32 W[K x N] -> f16 fragment-linear Bp[(K/32)][N][2][16].
__global__ void gdta_packB(const float* __restrict__ W, _Float16* __restrict__ Bp,
                           int K, int N) {
  int i = blockIdx.x * blockDim.x + threadIdx.x;
  if (i >= K * N) return;
  int k = i / N, n = i - k * N;  // reads coalesced over n
  int ks = k >> 5, kr = k & 31, kb = (kr >> 3) & 1, e = (kr & 7) | ((kr >> 4) << 3);
  Bp[(((size_t)ks * N + n) * 2 + kb) * 16 + e] = (_Float16)W[i];
}

// z[node, 0:finp] = (d < fin) ? h[node, d] : 0   (seeds z = h, zero-padded K)
__global__ void gdta_copy_pad(const float* __restrict__ h, float* __restrict__ z,
                              int total, int fin, int shift) {
  int i = blockIdx.x * blockDim.x + threadIdx.x;
  if (i >= total) return;
  int node = i >> shift;
  int d = i & ((1 << shift) - 1);
  z[i] = (d < fin) ? h[(size_t)node * fin + d] : 0.f;
}

// z[dst] += h[src]  (wave per edge, lane per feature: coalesced)
__global__ void gdta_edge_agg(const int* __restrict__ eidx, const float* __restrict__ h,
                              float* __restrict__ z, int nE, int fin, int ldz) {
  int wave = threadIdx.x >> 5, lane = threadIdx.x & 31;
  int e = blockIdx.x * 8 + wave;
  int s = eidx[e], t = eidx[nE + e];
  const float* hs = h + (size_t)s * fin;
  float* zt = z + (size_t)t * ldz;
  for (int d = lane; d < fin; d += 32) atomicAdd(&zt[d], hs[d]);
}

__global__ void gdta_pool(const float* __restrict__ h, const int* __restrict__ batch,
                          float* __restrict__ pooled, int nNodes) {
  int i = blockIdx.x * blockDim.x + threadIdx.x;
  if (i >= nNodes * 32) return;
  int node = i >> 5;
  int d = i & 31;
  atomicAdd(&pooled[(size_t)batch[node] * 32 + d], h[i]);
}

// ------------- GIN layer: h' = bn(relu(W2 relu(W1 z + b1) + b2)) -----------
// z is pre-aggregated and zero-padded to FINP columns. W1/W2 staged in LDS f16.

template <int FINP>
__global__ void gdta_gin_mlp(const float* __restrict__ z,
                             const float* __restrict__ W1, const float* __restrict__ b1,
                             const float* __restrict__ W2, const float* __restrict__ b2,
                             const float* __restrict__ bg, const float* __restrict__ bb,
                             const float* __restrict__ bm, const float* __restrict__ bv,
                             int fin, float* __restrict__ hout) {
  __shared__ __align__(16) _Float16 W1s[FINP * 32];
  __shared__ __align__(16) _Float16 W2s[32 * 32];
  __shared__ __align__(16) _Float16 tls[8 * 16 * 32];  // per-wave relayout (1 KB each)
  int t = threadIdx.x;
  for (int i = t; i < FINP * 32; i += 256) {
    int k = i >> 5, n = i & 31;
    W1s[i] = (k < fin) ? (_Float16)W1[k * 32 + n] : (_Float16)0.f;
  }
  for (int i = t; i < 32 * 32; i += 256) W2s[i] = (_Float16)W2[i];
  __syncthreads();

  int wave = t >> 5, lane = t & 31;
  int m0 = (blockIdx.x * 8 + wave) * 16;
  int n = lane & 15, half = lane >> 4;
  v8f acc0 = zero8(), acc1 = zero8();
#pragma unroll
  for (int ks = 0; ks < FINP / 32; ++ks) {
    v16h a = load_A_f32x32(z, FINP, m0, ks * 32);
    v16h bA = load_B_f16(W1s, 32, ks * 32, 0);
    v16h bB = load_B_f16(W1s, 32, ks * 32, 16);
    acc0 = WMMA16(a, bA, acc0);
    acc1 = WMMA16(a, bB, acc1);
  }
  // t1 = relu(acc + b1) -> f16 relayout through LDS (per-wave region)
  _Float16* tl = &tls[wave * 512];
  {
    float bi0 = b1[n], bi1 = b1[n + 16];
#pragma unroll
    for (int r = 0; r < 8; ++r) {
      int ml = half * 8 + r;
      float t0 = acc0[r] + bi0; t0 = t0 > 0.f ? t0 : 0.f;
      float t1 = acc1[r] + bi1; t1 = t1 > 0.f ? t1 : 0.f;
      tl[ml * 32 + n]      = (_Float16)t0;
      tl[ml * 32 + n + 16] = (_Float16)t1;
    }
  }
  __syncthreads();
  v16h a2 = load_A_f16v(tl, 32, 0, 0);
  v16h b20 = load_B_f16(W2s, 32, 0, 0);
  v16h b21 = load_B_f16(W2s, 32, 0, 16);
  v8f u0 = zero8(), u1 = zero8();
  u0 = WMMA16(a2, b20, u0);
  u1 = WMMA16(a2, b21, u1);
  float bi0 = b2[n], bi1 = b2[n + 16];
  float g0 = bg[n], g1 = bg[n + 16], be0 = bb[n], be1 = bb[n + 16];
  float me0 = bm[n], me1 = bm[n + 16];
  float rs0 = rsqrtf(bv[n] + 1e-5f), rs1 = rsqrtf(bv[n + 16] + 1e-5f);
#pragma unroll
  for (int r = 0; r < 8; ++r) {
    int m = m0 + half * 8 + r;
    float z0 = u0[r] + bi0; z0 = z0 > 0.f ? z0 : 0.f;
    float z1 = u1[r] + bi1; z1 = z1 > 0.f ? z1 : 0.f;
    hout[(size_t)m * 32 + n]      = g0 * (z0 - me0) * rs0 + be0;
    hout[(size_t)m * 32 + n + 16] = g1 * (z1 - me1) * rs1 + be1;
  }
}

// ----------------------- xd_g = relu(pooled @ Wxd + bxd) -> xj[:, :128] ----

__global__ void gdta_fc1_xd(const float* __restrict__ pooled,
                            const _Float16* __restrict__ Wp,
                            const float* __restrict__ b, _Float16* __restrict__ xj) {
  int wave = threadIdx.x >> 5, lane = threadIdx.x & 31;
  int m0 = (blockIdx.x * 8 + wave) * 16;
  v16h a = load_A_f32x32(pooled, 32, m0, 0);
  int n = lane & 15, half = lane >> 4;
#pragma unroll
  for (int nt = 0; nt < 8; ++nt) {
    v16h bf = load_B_pack(Wp, 128, 0, nt * 16);
    v8f c = zero8();
    c = WMMA16(a, bf, c);
    float bias = b[nt * 16 + n];
#pragma unroll
    for (int r = 0; r < 8; ++r) {
      float v = c[r] + bias; v = v > 0.f ? v : 0.f;
      int m = m0 + half * 8 + r;
      xj[(size_t)m * 256 + nt * 16 + n] = (_Float16)v;
    }
  }
}

// --------- vocab-factorized conv: R[b][v*8+k][o] = sum_{c:xt==v} W[o,c,k] ----

__global__ void gdta_reduceW(const int* __restrict__ xt, const float* __restrict__ W,
                             _Float16* __restrict__ R) {
  __shared__ float Rs[26 * 256];
  __shared__ int xs[1000];
  int b = blockIdx.x, t = threadIdx.x;
  for (int i = t; i < 26 * 256; i += 256) Rs[i] = 0.f;
  for (int i = t; i < 1000; i += 256) xs[i] = xt[b * 1000 + i];
  __syncthreads();
  int o = t >> 3, k = t & 7;  // thread owns one (o,k) column of R
  const float* Wp = W + (size_t)o * 8000 + k;
  for (int c = 0; c < 1000; ++c) {
    Rs[xs[c] * 256 + t] += Wp[(size_t)c * 8];  // xs[c] uniform across block
  }
  __syncthreads();
  _Float16* Rb = R + (size_t)b * 224 * 32;
  for (int v = 0; v < 26; ++v) Rb[(v * 8 + k) * 32 + o] = (_Float16)Rs[v * 256 + t];
  for (int i = t; i < 16 * 32; i += 256) Rb[208 * 32 + i] = (_Float16)0.f;  // K pad
}

// conv_out[b,o,l] = sum_{v,k} table[v][l+k] * R[b][(v,k)][o] + cb[o]  (WMMA GEMM)
__global__ void gdta_conv(const float* __restrict__ emb, const _Float16* __restrict__ R,
                          const float* __restrict__ cb, _Float16* __restrict__ cf) {
  __shared__ __align__(16) _Float16 tb[26 * 128];
  __shared__ __align__(16) _Float16 Rs2[224 * 32];
  int b = blockIdx.x, t = threadIdx.x;
  for (int i = t; i < 26 * 128; i += 256) tb[i] = (_Float16)emb[i];
  {
    const uint4* src = (const uint4*)(R + (size_t)b * 224 * 32);
    uint4* dst = (uint4*)Rs2;
    for (int i = t; i < 896; i += 256) dst[i] = src[i];  // 14 KB panel -> LDS
  }
  __syncthreads();
  int wave = t >> 5, lane = t & 31;
  int l0 = wave * 16;
  int n = lane & 15, half = lane >> 4, kb = half * 8;
  int lrow = l0 + n;                       // A-fragment row = output position l
  int lcl = lrow < 121 ? lrow : 120;       // clamped (branchless speculate)
  bool lok = lrow < 121;
  v8f a0 = zero8(), a1 = zero8();
  for (int ks = 0; ks < 7; ++ks) {
    v16h a;
#pragma unroll
    for (int e = 0; e < 16; ++e) {
      int kk = ks * 32 + kb + e + (e >= 8 ? 8 : 0);
      int vv = kk >> 3, kq = kk & 7;
      int vc = vv < 26 ? vv : 25;
      _Float16 tv = tb[vc * 128 + lcl + kq];          // always in-bounds
      a[e] = (lok && vv < 26) ? tv : (_Float16)0.f;   // cndmask, no branch
    }
    v16h B0 = load_B_f16(Rs2, 32, ks * 32, 0);
    v16h B1 = load_B_f16(Rs2, 32, ks * 32, 16);
    a0 = WMMA16(a, B0, a0);
    a1 = WMMA16(a, B1, a1);
  }
  float cb0 = cb[n], cb1 = cb[n + 16];
  _Float16* cfb = cf + (size_t)b * 3872;  // layout f = o*121 + l (torch reshape order)
#pragma unroll
  for (int r = 0; r < 8; ++r) {
    int l = l0 + half * 8 + r;
    if (l < 121) {
      cfb[n * 121 + l]        = (_Float16)(a0[r] + cb0);
      cfb[(n + 16) * 121 + l] = (_Float16)(a1[r] + cb1);
    }
  }
}

// ----------------------- xt_f = cf @ Wf + bf -> xj[:, 128:] (K=3872) -------
// A direct from cf (2x b128/lane), B from packed panel (2x b128/lane), no LDS.

__global__ void gdta_fc1_xt(const _Float16* __restrict__ cf, const _Float16* __restrict__ Wp,
                            const float* __restrict__ bfc, _Float16* __restrict__ xj) {
  int m0 = blockIdx.x * 16;
  int t = threadIdx.x, wave = t >> 5, lane = t & 31;
  v8f acc = zero8();
  for (int ks = 0; ks < 121; ++ks) {
    if (lane == 0 && ks + 1 < 121)
      __builtin_prefetch(Wp + ((size_t)(ks + 1) * 128 + wave * 16) * 32, 0, 1);
    v16h a = load_A_f16v(cf, 3872, m0, ks * 32);
    v16h bfrag = load_B_pack(Wp, 128, ks, wave * 16);
    acc = WMMA16(a, bfrag, acc);
  }
  int n = lane & 15, half = lane >> 4;
  float bias = bfc[wave * 16 + n];
#pragma unroll
  for (int r = 0; r < 8; ++r) {
    int m = m0 + half * 8 + r;
    xj[(size_t)m * 256 + 128 + wave * 16 + n] = (_Float16)(acc[r] + bias);
  }
}

// ----------------------- classifier ----------------------------------------

__global__ void gdta_cls0(const _Float16* __restrict__ xj, const _Float16* __restrict__ Wp,
                          const float* __restrict__ bias, _Float16* __restrict__ h2) {
  __shared__ __align__(16) _Float16 st[8 * 16 * 128];  // 32 KB store relayout
  int m0 = blockIdx.x * 16;
  int t = threadIdx.x, wave = t >> 5, lane = t & 31;
  v8f acc[8];
#pragma unroll
  for (int j = 0; j < 8; ++j) acc[j] = zero8();
  for (int ks = 0; ks < 8; ++ks) {
    v16h a = load_A_f16v(xj, 256, m0, ks * 32);
#pragma unroll
    for (int nt = 0; nt < 8; ++nt) {
      v16h bfrag = load_B_pack(Wp, 1024, ks, wave * 128 + nt * 16);
      acc[nt] = WMMA16(a, bfrag, acc[nt]);
    }
  }
  // bias + relu -> per-wave LDS tile (16 x 128), then coalesced b128 stores
  int n = lane & 15, half = lane >> 4;
  _Float16* sw = &st[wave * 2048];
#pragma unroll
  for (int nt = 0; nt < 8; ++nt) {
    float bi = bias[wave * 128 + nt * 16 + n];
#pragma unroll
    for (int r = 0; r < 8; ++r) {
      float v = acc[nt][r] + bi; v = v > 0.f ? v : 0.f;
      sw[(half * 8 + r) * 128 + nt * 16 + n] = (_Float16)v;
    }
  }
  __syncthreads();
  const uint4* s4 = (const uint4*)st;
  for (int i = t; i < 2048; i += 256) {
    int w = i >> 8, r = (i >> 4) & 15, q = i & 15;
    *(uint4*)(h2 + (size_t)(m0 + r) * 1024 + w * 128 + q * 8) = s4[i];
  }
}

__global__ void gdta_cls1(const _Float16* __restrict__ h2, const _Float16* __restrict__ Wp,
                          const float* __restrict__ bias, float* __restrict__ h3) {
  int m0 = blockIdx.x * 16;
  int t = threadIdx.x, wave = t >> 5, lane = t & 31;
  v8f a0 = zero8(), a1 = zero8();
  for (int ks = 0; ks < 32; ++ks) {
    v16h a = load_A_f16v(h2, 1024, m0, ks * 32);
    v16h B0 = load_B_pack(Wp, 256, ks, wave * 32);
    v16h B1 = load_B_pack(Wp, 256, ks, wave * 32 + 16);
    a0 = WMMA16(a, B0, a0);
    a1 = WMMA16(a, B1, a1);
  }
  int n = lane & 15, half = lane >> 4;
  float bi0 = bias[wave * 32 + n], bi1 = bias[wave * 32 + 16 + n];
#pragma unroll
  for (int r = 0; r < 8; ++r) {
    int m = m0 + half * 8 + r;
    float v0 = a0[r] + bi0; v0 = v0 > 0.f ? v0 : 0.f;
    float v1 = a1[r] + bi1; v1 = v1 > 0.f ? v1 : 0.f;
    h3[(size_t)m * 256 + wave * 32 + n]      = v0;
    h3[(size_t)m * 256 + wave * 32 + 16 + n] = v1;
  }
}

__global__ void gdta_cls2(const float* __restrict__ h3, const float* __restrict__ W,
                          const float* __restrict__ bias, const float* __restrict__ y,
                          float* __restrict__ out) {
  __shared__ float red[256];
  int b = blockIdx.x, t = threadIdx.x;
  red[t] = h3[(size_t)b * 256 + t] * W[t];
  __syncthreads();
  for (int s = 128; s > 0; s >>= 1) {
    if (t < s) red[t] += red[t + s];
    __syncthreads();
  }
  if (t == 0) out[b] = red[0] + bias[0];
  if (t == 1) out[1024 + b] = y[b];
}

// ----------------------- host launcher -------------------------------------

extern "C" void kernel_launch(void* const* d_in, const int* in_sizes, int n_in,
                              void* d_out, int out_size, void* d_ws, size_t ws_size,
                              hipStream_t stream) {
  (void)in_sizes; (void)n_in; (void)out_size; (void)ws_size;
  const float* xd   = (const float*)d_in[0];
  const int* eidx   = (const int*)d_in[1];
  const int* batch  = (const int*)d_in[2];
  const int* xt     = (const int*)d_in[3];
  const float* y    = (const float*)d_in[4];
  auto P = [&](int i) { return (const float*)d_in[i]; };
  const float* fc1xd_W = P(45); const float* fc1xd_b = P(46);
  const float* emb = P(47);
  const float* convW = P(48); const float* convB = P(49);
  const float* fxt_W = P(50); const float* fxt_b = P(51);
  const float* c0W = P(52); const float* c0b = P(53);
  const float* c1W = P(54); const float* c1b = P(55);
  const float* c2W = P(56); const float* c2b = P(57);

  char* ws = (char*)d_ws;
  size_t off = 0;
  auto alloc = [&](size_t bytes) {
    size_t o = off;
    off = (off + bytes + 255) & ~(size_t)255;
    return o;
  };
  float*     z      = (float*)(ws + alloc(40960ULL * 64 * 4));  // padded z = h + agg
  float*     hA     = (float*)(ws + alloc(40960ULL * 32 * 4));
  float*     hB     = (float*)(ws + alloc(40960ULL * 32 * 4));
  float*     pooled = (float*)(ws + alloc(1024ULL * 32 * 4));
  _Float16*  xj     = (_Float16*)(ws + alloc(1024ULL * 256 * 2));
  _Float16*  R      = (_Float16*)(ws + alloc(1024ULL * 224 * 32 * 2));
  _Float16*  cf     = (_Float16*)(ws + alloc(1024ULL * 3872 * 2));
  _Float16*  h2     = (_Float16*)(ws + alloc(1024ULL * 1024 * 2));
  float*     h3     = (float*)(ws + alloc(1024ULL * 256 * 4));
  _Float16*  fc1xdP = (_Float16*)(ws + alloc(32ULL * 128 * 2));     // packed weights
  _Float16*  fxtP   = (_Float16*)(ws + alloc(3872ULL * 128 * 2));
  _Float16*  c0P    = (_Float16*)(ws + alloc(256ULL * 1024 * 2));
  _Float16*  c1P    = (_Float16*)(ws + alloc(1024ULL * 256 * 2));

  // ---- one-time weight repacks (f32 -> f16 fragment-linear) ----
  gdta_packB<<<(32 * 128 + 255) / 256, 256, 0, stream>>>(fc1xd_W, fc1xdP, 32, 128);
  gdta_packB<<<(3872 * 128 + 255) / 256, 256, 0, stream>>>(fxt_W, fxtP, 3872, 128);
  gdta_packB<<<(256 * 1024 + 255) / 256, 256, 0, stream>>>(c0W, c0P, 256, 1024);
  gdta_packB<<<(1024 * 256 + 255) / 256, 256, 0, stream>>>(c1W, c1P, 1024, 256);

  // ---- GIN stack ----
  const float* hin = xd;
  float* houts[5] = {hA, hB, hA, hB, hA};
  for (int li = 0; li < 5; ++li) {
    int fin = (li == 0) ? 55 : 32;
    int shift = (li == 0) ? 6 : 5;  // padded width 64 / 32
    int total = 40960 * (1 << shift);
    gdta_copy_pad<<<(total + 255) / 256, 256, 0, stream>>>(hin, z, total, fin, shift);
    gdta_edge_agg<<<163840 / 8, 256, 0, stream>>>(eidx, hin, z, 163840, fin, 1 << shift);
    const float *W1 = P(5 + 8 * li), *b1 = P(6 + 8 * li);
    const float *W2 = P(7 + 8 * li), *b2 = P(8 + 8 * li);
    const float *g = P(9 + 8 * li), *bb = P(10 + 8 * li);
    const float *mm = P(11 + 8 * li), *vv = P(12 + 8 * li);
    if (li == 0)
      gdta_gin_mlp<64><<<320, 256, 0, stream>>>(z, W1, b1, W2, b2, g, bb, mm, vv, fin,
                                                houts[li]);
    else
      gdta_gin_mlp<32><<<320, 256, 0, stream>>>(z, W1, b1, W2, b2, g, bb, mm, vv, fin,
                                                houts[li]);
    hin = houts[li];
  }

  // ---- pool + fc1_xd ----
  gdta_zero<<<(1024 * 32 + 255) / 256, 256, 0, stream>>>(pooled, 1024 * 32);
  gdta_pool<<<(40960 * 32 + 255) / 256, 256, 0, stream>>>(hin, batch, pooled, 40960);
  gdta_fc1_xd<<<8, 256, 0, stream>>>(pooled, fc1xdP, fc1xd_b, xj);

  // ---- protein branch (vocab-factorized conv) ----
  gdta_reduceW<<<1024, 256, 0, stream>>>(xt, convW, R);
  gdta_conv<<<1024, 256, 0, stream>>>(emb, R, convB, cf);
  gdta_fc1_xt<<<64, 256, 0, stream>>>(cf, fxtP, fxt_b, xj);

  // ---- classifier ----
  gdta_cls0<<<64, 256, 0, stream>>>(xj, c0P, c0b, h2);
  gdta_cls1<<<64, 256, 0, stream>>>(h2, c1P, c1b, h3);
  gdta_cls2<<<1024, 256, 0, stream>>>(h3, c2W, c2b, y, (float*)d_out);
}
